// TCN_GCN_unit_3762391351396
// MI455X (gfx1250) — matrix operand order repaired and checked by reading, not compiled, not run on hardware
//
#include <hip/hip_runtime.h>
#include <math.h>

// ---------------- types ----------------
typedef __attribute__((ext_vector_type(16))) __bf16 v16bf;
typedef __attribute__((ext_vector_type(8)))  float  v8f;

union Frag { unsigned int u[8]; v16bf v; };

__device__ __forceinline__ unsigned short f2bf(float f) {
    unsigned int u = __builtin_bit_cast(unsigned int, f);
    u += 0x7fffu + ((u >> 16) & 1u);
    return (unsigned short)(u >> 16);
}

// dims
#define NB   32
#define CIN  64
#define TT   256
#define VV   25
#define COUT 192
#define INTER 48
#define PIX  6400          // T*V per n
#define PTOT 204800        // N*T*V
#define LP   40            // LDS row pitch in bf16 (80B: 16B-aligned rows for async B128)

// ---------------------------------------------------------------------------
// x second-moment partials: per block (512 pixels) -> M2 [64x64], mean [64]
// ---------------------------------------------------------------------------
__global__ __launch_bounds__(256) void k_xstats(const float* __restrict__ x,
                                                double* __restrict__ meanp,   // [400][64]
                                                double* __restrict__ m2p)     // [400][4096]
{
    const int tid = threadIdx.x;
    const long pstart = (long)blockIdx.x * 512;
    __shared__ float xs[64 * 128];
    float accm[16];
#pragma unroll
    for (int i = 0; i < 16; ++i) accm[i] = 0.f;
    float accs = 0.f;

    for (int chunk = 0; chunk < 4; ++chunk) {
        long pb = pstart + chunk * 128;
        __syncthreads();
        for (int i = 0; i < 32; ++i) {
            int idx = tid + i * 256;            // 0..8191
            int c = idx >> 7, p = idx & 127;
            long gp = pb + p;
            int nn = (int)(gp / PIX);
            int off = (int)(gp % PIX);
            xs[c * 128 + p] = x[((long)nn * CIN + c) * PIX + off];
        }
        __syncthreads();
#pragma unroll
        for (int i = 0; i < 16; ++i) {
            int pair = tid * 16 + i;            // 0..4095
            int c1 = pair >> 6, c2 = pair & 63;
            float a = 0.f;
            for (int p = 0; p < 128; ++p) a += xs[c1 * 128 + p] * xs[c2 * 128 + p];
            accm[i] += a;
        }
        if (tid < 64) {
            float a = 0.f;
            for (int p = 0; p < 128; ++p) a += xs[tid * 128 + p];
            accs += a;
        }
    }
#pragma unroll
    for (int i = 0; i < 16; ++i) m2p[(long)blockIdx.x * 4096 + tid * 16 + i] = (double)accm[i];
    if (tid < 64) meanp[(long)blockIdx.x * 64 + tid] = (double)accs;
}

__global__ __launch_bounds__(256) void k_xred(const double* __restrict__ meanp,
                                              const double* __restrict__ m2p,
                                              double* __restrict__ meanx,   // [64]
                                              double* __restrict__ M2)      // [4096]
{
    int i = blockIdx.x * 256 + threadIdx.x;
    if (i < 4096) {
        double s = 0.0;
        for (int b = 0; b < 400; ++b) s += m2p[(long)b * 4096 + i];
        M2[i] = s;
    } else if (i < 4160) {
        int c = i - 4096;
        double s = 0.0;
        for (int b = 0; b < 400; ++b) s += meanp[(long)b * 64 + c];
        meanx[c] = s;
    }
}

// ---------------------------------------------------------------------------
// weight prep: Wd_flat bf16, bdsum, 11 fused TCN taps (bf16), tap biases
// ---------------------------------------------------------------------------
__global__ __launch_bounds__(256) void k_prep(const float* __restrict__ Wd,  // [3][192][64]
                                              const float* __restrict__ bd,  // [3][192]
                                              const float* __restrict__ W11, const float* __restrict__ b11,
                                              const float* __restrict__ W21, const float* __restrict__ b21,
                                              const float* __restrict__ W31, const float* __restrict__ b31,
                                              const float* __restrict__ W12, const float* __restrict__ b12,
                                              const float* __restrict__ W22, const float* __restrict__ b22,
                                              const float* __restrict__ W32, const float* __restrict__ b32,
                                              unsigned short* __restrict__ Wdf,   // [192][192] bf16
                                              unsigned short* __restrict__ U,     // [11][64][192] bf16
                                              float* __restrict__ bdsum,          // [192]
                                              float* __restrict__ cb_tap)         // [192]
{
    const int tid = threadIdx.x;
    if (blockIdx.x == 0) {
        for (int idx = tid; idx < COUT * COUT; idx += 256) {
            int o = idx / 192, kc = idx % 192;
            int k = kc / 64, c = kc % 64;
            Wdf[o * 192 + kc] = f2bf(Wd[((long)k * 192 + o) * 64 + c]);
        }
        if (tid < 192) bdsum[tid] = bd[tid] + bd[192 + tid] + bd[384 + tid];
        if (tid < 192) {
            int br = tid / 64, m = tid % 64;
            float a;
            if (br == 0) {
                a = b12[m];
                for (int j = 0; j < 192; ++j) a += W12[m * 192 + j] * b11[j];
            } else if (br == 1) {
                a = b22[m];
                for (int jj = 0; jj < 3; ++jj)
                    for (int j = 0; j < 192; ++j) a += W22[(m * 192 + j) * 3 + jj] * b21[j];
            } else {
                a = b32[m];
                for (int jj = 0; jj < 7; ++jj)
                    for (int j = 0; j < 192; ++j) a += W32[(m * 192 + j) * 7 + jj] * b31[j];
            }
            cb_tap[tid] = a;
        }
    } else {
        int tap = blockIdx.x - 1;   // 0..10
        for (int idx = tid; idx < 64 * 192; idx += 256) {
            int m = idx / 192, c = idx % 192;
            float a = 0.f;
            if (tap == 0) {
                for (int j = 0; j < 192; ++j) a += W12[m * 192 + j] * W11[j * 192 + c];
            } else if (tap <= 3) {
                int jj = tap - 1;
                for (int j = 0; j < 192; ++j) a += W22[(m * 192 + j) * 3 + jj] * W21[j * 192 + c];
            } else {
                int jj = tap - 4;
                for (int j = 0; j < 192; ++j) a += W32[(m * 192 + j) * 7 + jj] * W31[j * 192 + c];
            }
            U[((long)tap * 64 + m) * 192 + c] = f2bf(a);
        }
    }
}

// ---------------------------------------------------------------------------
// analytic BN for down/res paths; writes BN-scaled bf16 weights + fused biases
// ---------------------------------------------------------------------------
__global__ __launch_bounds__(256) void k_bn_lin(const double* __restrict__ meanx, const double* __restrict__ M2,
                                                const float* __restrict__ Wdn, const float* __restrict__ bdn,
                                                const float* __restrict__ gdn, const float* __restrict__ betadn,
                                                const float* __restrict__ Wrs, const float* __restrict__ brs,
                                                const float* __restrict__ grs, const float* __restrict__ betars,
                                                unsigned short* __restrict__ Wdbf, unsigned short* __restrict__ Wrbf,
                                                float* __restrict__ cbias_g, float* __restrict__ cbias_fin)
{
    __shared__ double mx[64];
    __shared__ double m2s[4096];
    const int tid = threadIdx.x;
    const double P = (double)PTOT;
    for (int i = tid; i < 64; i += 256) mx[i] = meanx[i] / P;
    for (int i = tid; i < 4096; i += 256) m2s[i] = M2[i] / P;
    __syncthreads();
    for (int r = tid; r < 384; r += 256) {
        int path = r / 192, o = r % 192;
        const float* W = path ? Wrs : Wdn;
        const float* bb = path ? brs : bdn;
        const float* ga = path ? grs : gdn;
        const float* be = path ? betars : betadn;
        double mw = 0.0;
        for (int c = 0; c < 64; ++c) mw += (double)W[o * 64 + c] * mx[c];
        double e2 = 0.0;
        for (int c1 = 0; c1 < 64; ++c1) {
            double t = 0.0;
            for (int c2 = 0; c2 < 64; ++c2) t += (double)W[o * 64 + c2] * m2s[c1 * 64 + c2];
            e2 += (double)W[o * 64 + c1] * t;
        }
        double bi = (double)bb[o];
        double mean = mw + bi;
        double E2 = e2 + 2.0 * bi * mw + bi * bi;
        double var = E2 - mean * mean;
        double sc = (double)ga[o] / sqrt(var + 1e-5);
        double sh = (double)be[o] - mean * sc;
        unsigned short* Wout = path ? Wrbf : Wdbf;
        for (int c = 0; c < 64; ++c) Wout[o * 64 + c] = f2bf((float)(sc * (double)W[o * 64 + c]));
        float* cb = path ? cbias_fin : cbias_g;
        cb[o] = (float)(sc * bi + sh);
    }
}

// ---------------------------------------------------------------------------
// attention: per (n,k,t-chunk) accumulate 25x25 score partials
// ---------------------------------------------------------------------------
__global__ __launch_bounds__(256) void k_att_scores(const float* __restrict__ x,
                                                    const float* __restrict__ Wa, const float* __restrict__ ba,
                                                    const float* __restrict__ Wb, const float* __restrict__ bb,
                                                    float* __restrict__ scp)     // [8][96][625]
{
    const int nk = blockIdx.x;              // 0..95
    const int tch = blockIdx.y;             // 0..7
    const int n = nk / 3, k = nk % 3;
    const int tid = threadIdx.x;
    __shared__ float WaL[3072], WbL[3072], xt[1600], fa[1200], fb[1200], sc[625];
    for (int i = tid; i < 3072; i += 256) { WaL[i] = Wa[k * 3072 + i]; WbL[i] = Wb[k * 3072 + i]; }
    for (int i = tid; i < 625; i += 256) sc[i] = 0.f;
    __syncthreads();
    for (int t = tch * 32; t < tch * 32 + 32; ++t) {
        __syncthreads();
        for (int i = tid; i < 1600; i += 256) {
            int c = i / 25, v = i % 25;
            xt[i] = x[(((long)n * CIN + c) * TT + t) * VV + v];
        }
        __syncthreads();
        for (int i = tid; i < 2400; i += 256) {
            int which = i >= 1200;
            int e = which ? i - 1200 : i;
            int ii = e / 25, v = e % 25;
            const float* W = which ? WbL : WaL;
            float a = which ? bb[k * 48 + ii] : ba[k * 48 + ii];
            for (int c = 0; c < 64; ++c) a += W[ii * 64 + c] * xt[c * 25 + v];
            if (which) fb[e] = a; else fa[e] = a;
        }
        __syncthreads();
        for (int e = tid; e < 625; e += 256) {
            int v = e / 25, w = e % 25;
            float a = 0.f;
            for (int ii = 0; ii < 48; ++ii) a += fa[ii * 25 + v] * fb[ii * 25 + w];
            sc[e] += a;
        }
    }
    __syncthreads();
    for (int e = tid; e < 625; e += 256) scp[((long)tch * 96 + nk) * 625 + e] = sc[e];
}

__global__ void k_softmax(const float* __restrict__ scp, const float* __restrict__ A,
                          const float* __restrict__ PA, float* __restrict__ att)
{
    const int nk = blockIdx.x;
    const int k = nk % 3;
    const int w = threadIdx.x;
    if (w >= 25) return;
    float col[25];
    float mx = -3.4e38f;
    for (int v = 0; v < 25; ++v) {
        float s = 0.f;
        for (int tch = 0; tch < 8; ++tch) s += scp[((long)tch * 96 + nk) * 625 + v * 25 + w];
        col[v] = s * (1.f / 12288.f);
        mx = col[v] > mx ? col[v] : mx;
    }
    float ssum = 0.f;
    for (int v = 0; v < 25; ++v) { col[v] = __expf(col[v] - mx); ssum += col[v]; }
    float inv = 1.f / ssum;
    for (int v = 0; v < 25; ++v)
        att[(long)nk * 625 + v * 25 + w] = col[v] * inv + A[k * 625 + v * 25 + w] + PA[k * 625 + v * 25 + w];
}

// ---------------------------------------------------------------------------
// z' = x @ att  -> bf16, PIXEL-MAJOR: Xp[n][p][kc]  (kc = k*64+c, 192 per pixel)
// ---------------------------------------------------------------------------
__global__ __launch_bounds__(256) void k_zprime(const float* __restrict__ x,
                                                const float* __restrict__ att,
                                                unsigned short* __restrict__ Xp)
{
    const int tch = blockIdx.x;   // 0..7
    const int k = blockIdx.y;     // 0..2
    const int n = blockIdx.z;     // 0..31
    const int tid = threadIdx.x;
    __shared__ float attL[625];
    for (int i = tid; i < 625; i += 256) attL[i] = att[((long)n * 3 + k) * 625 + i];
    __syncthreads();
    for (int r = tid; r < 2048; r += 256) {
        int c = r >> 5;
        int t = tch * 32 + (r & 31);
        const float* xr = &x[(((long)n * CIN + c) * TT + t) * VV];
        float xl[25];
#pragma unroll
        for (int v = 0; v < 25; ++v) xl[v] = xr[v];
        unsigned short* op = &Xp[((long)n * PIX + t * 25) * COUT + k * 64 + c];
#pragma unroll
        for (int w = 0; w < 25; ++w) {
            float a = 0.f;
#pragma unroll
            for (int v = 0; v < 25; ++v) a += xl[v] * attL[v * 25 + w];
            op[(long)w * COUT] = f2bf(a);
        }
    }
}

// ---------------------------------------------------------------------------
// generic 192xP GEMM via v_wmma_f32_16x16x32_bf16
//  XBF16=1: X is bf16, PIXEL-major [n][p][Kdim] -> async global->LDS staging
//  XBF16=0: X is f32, CHANNEL-major [n][k][p]   -> VALU transpose staging
//  out: OUTBF16=1 -> bf16 pixel-major [n][p][192]; else f32 channel-major
// ---------------------------------------------------------------------------
template <int XBF16, int OUTBF16, int HASADD, int RELU>
__global__ __launch_bounds__(256) void k_gemm(const unsigned short* __restrict__ Wbf,
                                              const void* __restrict__ Xall, long x_nstride,
                                              const float* __restrict__ bias,
                                              const float* __restrict__ addend,
                                              const float* __restrict__ ascale,
                                              const float* __restrict__ aoff,
                                              void* __restrict__ out, int Kdim)
{
    const int n = blockIdx.y;
    const int pblock = blockIdx.x * 128;
    const int tid = threadIdx.x;
    const int wave = tid >> 5, lane = tid & 31;
    const int nn = lane & 15, half = lane >> 4;

    __shared__ unsigned short Xs[128 * LP];   // pixel-major, pitch 40 shorts (80B rows)

    v8f acc[12];
#pragma unroll
    for (int mt = 0; mt < 12; ++mt)
#pragma unroll
        for (int r = 0; r < 8; ++r) acc[mt][r] = 0.f;

    const float* Xf = (const float*)Xall + (long)n * x_nstride;
    const unsigned short* Xb = (const unsigned short*)Xall + (long)n * x_nstride;
    const unsigned xs_base = (unsigned)(unsigned long long)(&Xs[0]);

    int k0a[8], k0b[8];
#pragma unroll
    for (int j = 0; j < 8; ++j) {
        k0a[j] = ((j >> 2) << 4) + ((j & 3) << 1) + (half << 3);   // A-frag K (16-bit A layout)
        k0b[j] = ((j >> 2) << 4) + ((j & 3) << 1) + (half << 3);   // B-frag K (16-bit B layout)
    }

    for (int kk = 0; kk < Kdim; kk += 32) {
        __syncthreads();
        if (XBF16) {
            // async global->LDS: 128 pixel rows x 64B (four 16B segments per row)
#pragma unroll
            for (int i = 0; i < 2; ++i) {
                int idx = tid + i * 256;      // 0..511
                int p = idx >> 2, seg = idx & 3;
                unsigned long long ga = (unsigned long long)(const void*)(Xb + (long)(pblock + p) * Kdim + kk)
                                        + (unsigned)(seg * 16);
                unsigned ldsa = xs_base + (unsigned)(p * (LP * 2) + seg * 16);
                asm volatile("global_load_async_to_lds_b128 %0, %1, off"
                             :: "v"(ldsa), "v"(ga) : "memory");
            }
            asm volatile("s_wait_asynccnt 0x0" ::: "memory");
        } else {
            // f32 channel-major source: convert + transpose into LDS
#pragma unroll
            for (int i = 0; i < 16; ++i) {
                int idx = tid + i * 256;      // 0..4095
                int k = idx >> 7, p = idx & 127;
                Xs[p * LP + k] = f2bf(Xf[(long)(kk + k) * PIX + pblock + p]);
            }
        }
        __syncthreads();
        Frag bfr;
#pragma unroll
        for (int j = 0; j < 8; ++j)
            bfr.u[j] = *(const unsigned int*)&Xs[(wave * 16 + nn) * LP + k0b[j]];
#pragma unroll
        for (int mt = 0; mt < 12; ++mt) {
            Frag afr;
            int m = mt * 16 + nn;
#pragma unroll
            for (int j = 0; j < 8; ++j)
                afr.u[j] = *(const unsigned int*)&Wbf[(long)m * Kdim + kk + k0a[j]];
            acc[mt] = __builtin_amdgcn_wmma_f32_16x16x32_bf16(false, afr.v, false, bfr.v,
                                                              (short)0, acc[mt], false, false);
        }
    }

    const int p = pblock + wave * 16 + nn;
    if (OUTBF16) {
        unsigned short* ob = (unsigned short*)out;
#pragma unroll
        for (int mt = 0; mt < 12; ++mt) {
            unsigned int pk[4];
#pragma unroll
            for (int e = 0; e < 4; ++e) {
                int m0 = mt * 16 + 2 * e + half * 8;
                float v0 = acc[mt][2 * e]     + bias[m0];
                float v1 = acc[mt][2 * e + 1] + bias[m0 + 1];
                if (HASADD) {
                    v0 += addend[((long)n * COUT + m0) * PIX + p] * ascale[m0] + aoff[m0];
                    v1 += addend[((long)n * COUT + m0 + 1) * PIX + p] * ascale[m0 + 1] + aoff[m0 + 1];
                }
                if (RELU) { v0 = v0 > 0.f ? v0 : 0.f; v1 = v1 > 0.f ? v1 : 0.f; }
                pk[e] = (unsigned)f2bf(v0) | ((unsigned)f2bf(v1) << 16);
            }
            uint4 q4; q4.x = pk[0]; q4.y = pk[1]; q4.z = pk[2]; q4.w = pk[3];
            *(uint4*)&ob[((long)n * PIX + p) * COUT + mt * 16 + half * 8] = q4;   // 16B contiguous
        }
    } else {
        float* of = (float*)out;
#pragma unroll
        for (int mt = 0; mt < 12; ++mt) {
#pragma unroll
            for (int r = 0; r < 8; ++r) {
                int m = mt * 16 + r + half * 8;
                float v = acc[mt][r] + bias[m];
                long o = ((long)n * COUT + m) * PIX + p;
                if (HASADD) v += addend[o] * ascale[m] + aoff[m];
                if (RELU) v = v > 0.f ? v : 0.f;
                of[o] = v;
            }
        }
    }
}

// ---------------------------------------------------------------------------
// fused TCN: 11 temporal taps over g (bf16 pixel-major [n][p][192]) -> tm (f32)
// ---------------------------------------------------------------------------
__global__ __launch_bounds__(256) void k_tcn(const unsigned short* __restrict__ U,   // [11][64][192]
                                             const unsigned short* __restrict__ G,   // [32][6400][192]
                                             const float* __restrict__ cbias,
                                             float* __restrict__ out)
{
    const int n = blockIdx.y;
    const int pblock = blockIdx.x * 128;
    const int tid = threadIdx.x;
    const int wave = tid >> 5, lane = tid & 31;
    const int nn = lane & 15, half = lane >> 4;

    __shared__ unsigned short Gs[278 * LP];   // pixels [pblock-75, pblock+203)

    v8f acc[12];
#pragma unroll
    for (int mt = 0; mt < 12; ++mt)
#pragma unroll
        for (int r = 0; r < 8; ++r) acc[mt][r] = 0.f;

    const unsigned short* Gn = G + (long)n * PIX * COUT;
    const unsigned gs_base = (unsigned)(unsigned long long)(&Gs[0]);

    int k0a[8], k0b[8];
#pragma unroll
    for (int j = 0; j < 8; ++j) {
        k0a[j] = ((j >> 2) << 4) + ((j & 3) << 1) + (half << 3);
        k0b[j] = ((j >> 2) << 4) + ((j & 3) << 1) + (half << 3);
    }

    for (int kk = 0; kk < 192; kk += 32) {
        __syncthreads();
        // async stage: 278 pixel rows x 64B; zero-fill halo rows outside [0,PIX)
#pragma unroll
        for (int i = 0; i < 5; ++i) {
            int idx = tid + i * 256;
            if (idx < 278 * 4) {
                int p = idx >> 2, seg = idx & 3;
                int q = pblock - 75 + p;          // shift by whole t rows (25 px per dt)
                if (q >= 0 && q < PIX) {
                    unsigned long long ga = (unsigned long long)(const void*)(Gn + (long)q * COUT + kk)
                                            + (unsigned)(seg * 16);
                    unsigned ldsa = gs_base + (unsigned)(p * (LP * 2) + seg * 16);
                    asm volatile("global_load_async_to_lds_b128 %0, %1, off"
                                 :: "v"(ldsa), "v"(ga) : "memory");
                } else {
                    unsigned long long* z = (unsigned long long*)&Gs[p * LP + seg * 8];
                    z[0] = 0ull; z[1] = 0ull;
                }
            }
        }
        asm volatile("s_wait_asynccnt 0x0" ::: "memory");
        __syncthreads();
        const int pl0 = wave * 16 + nn + 75;

        // branch 0: channels 0-63, single tap (dt=0), U tap 0
        {
            Frag bfr;
#pragma unroll
            for (int j = 0; j < 8; ++j)
                bfr.u[j] = *(const unsigned int*)&Gs[pl0 * LP + k0b[j]];
#pragma unroll
            for (int mt = 0; mt < 4; ++mt) {
                Frag afr;
                int mrow = mt * 16 + nn;
#pragma unroll
                for (int j = 0; j < 8; ++j)
                    afr.u[j] = *(const unsigned int*)&U[((long)0 * 64 + mrow) * 192 + kk + k0a[j]];
                acc[mt] = __builtin_amdgcn_wmma_f32_16x16x32_bf16(false, afr.v, false, bfr.v,
                                                                  (short)0, acc[mt], false, false);
            }
        }
        // branch 1: channels 64-127, taps dt=-1..1, U taps 1..3
#pragma unroll
        for (int j2 = 0; j2 < 3; ++j2) {
            int dt = j2 - 1;
            Frag bfr;
#pragma unroll
            for (int j = 0; j < 8; ++j)
                bfr.u[j] = *(const unsigned int*)&Gs[(pl0 + 25 * dt) * LP + k0b[j]];
#pragma unroll
            for (int mt = 4; mt < 8; ++mt) {
                Frag afr;
                int mrow = (mt - 4) * 16 + nn;
#pragma unroll
                for (int j = 0; j < 8; ++j)
                    afr.u[j] = *(const unsigned int*)&U[((long)(1 + j2) * 64 + mrow) * 192 + kk + k0a[j]];
                acc[mt] = __builtin_amdgcn_wmma_f32_16x16x32_bf16(false, afr.v, false, bfr.v,
                                                                  (short)0, acc[mt], false, false);
            }
        }
        // branch 2: channels 128-191, taps dt=-3..3, U taps 4..10
#pragma unroll
        for (int j3 = 0; j3 < 7; ++j3) {
            int dt = j3 - 3;
            Frag bfr;
#pragma unroll
            for (int j = 0; j < 8; ++j)
                bfr.u[j] = *(const unsigned int*)&Gs[(pl0 + 25 * dt) * LP + k0b[j]];
#pragma unroll
            for (int mt = 8; mt < 12; ++mt) {
                Frag afr;
                int mrow = (mt - 8) * 16 + nn;
#pragma unroll
                for (int j = 0; j < 8; ++j)
                    afr.u[j] = *(const unsigned int*)&U[((long)(4 + j3) * 64 + mrow) * 192 + kk + k0a[j]];
                acc[mt] = __builtin_amdgcn_wmma_f32_16x16x32_bf16(false, afr.v, false, bfr.v,
                                                                  (short)0, acc[mt], false, false);
            }
        }
    }

    const int p = pblock + wave * 16 + nn;
#pragma unroll
    for (int mt = 0; mt < 12; ++mt) {
#pragma unroll
        for (int r = 0; r < 8; ++r) {
            int m = mt * 16 + r + half * 8;
            out[((long)n * COUT + m) * PIX + p] = acc[mt][r] + cbias[m];
        }
    }
}

// ---------------------------------------------------------------------------
// per-channel sum/sumsq partials (deterministic, no atomics): [2][192][32]
// ---------------------------------------------------------------------------
__global__ __launch_bounds__(256) void k_stats(const float* __restrict__ buf, double* __restrict__ part)
{
    const int m = blockIdx.x, n = blockIdx.y, tid = threadIdx.x;
    const float* p = buf + ((long)n * COUT + m) * PIX;
    float s = 0.f, q = 0.f;
    for (int i = tid; i < PIX; i += 256) { float v = p[i]; s += v; q += v * v; }
    __shared__ float rs[256], rq[256];
    rs[tid] = s; rq[tid] = q;
    __syncthreads();
    for (int st = 128; st > 0; st >>= 1) {
        if (tid < st) { rs[tid] += rs[tid + st]; rq[tid] += rq[tid + st]; }
        __syncthreads();
    }
    if (tid == 0) {
        part[((long)0 * COUT + m) * 32 + n] = (double)rs[0];
        part[((long)1 * COUT + m) * 32 + n] = (double)rq[0];
    }
}

__global__ void k_bnfin(const double* __restrict__ part, const float* __restrict__ gamma,
                        const float* __restrict__ beta, float* __restrict__ s, float* __restrict__ t)
{
    int m = blockIdx.x * blockDim.x + threadIdx.x;
    if (m >= COUT) return;
    double sum = 0.0, sq = 0.0;
    for (int n = 0; n < 32; ++n) { sum += part[((long)0 * COUT + m) * 32 + n]; sq += part[((long)1 * COUT + m) * 32 + n]; }
    const double P = (double)PTOT;
    double mean = sum / P;
    double var = sq / P - mean * mean;
    double sc = (double)gamma[m] / sqrt(var + 1e-5);
    s[m] = (float)sc;
    t[m] = (float)((double)beta[m] - mean * sc);
}

// ---------------------------------------------------------------------------
extern "C" void kernel_launch(void* const* d_in, const int* in_sizes, int n_in,
                              void* d_out, int out_size, void* d_ws, size_t ws_size,
                              hipStream_t stream)
{
    const float* x    = (const float*)d_in[0];
    const float* A    = (const float*)d_in[1];
    const float* PA   = (const float*)d_in[2];
    const float* Wa   = (const float*)d_in[3];
    const float* ba   = (const float*)d_in[4];
    const float* Wb   = (const float*)d_in[5];
    const float* bbp  = (const float*)d_in[6];
    const float* Wd   = (const float*)d_in[7];
    const float* bd   = (const float*)d_in[8];
    const float* gcng = (const float*)d_in[9];
    const float* gcnb = (const float*)d_in[10];
    const float* dnw  = (const float*)d_in[11];
    const float* dnb  = (const float*)d_in[12];
    const float* dng  = (const float*)d_in[13];
    const float* dnbb = (const float*)d_in[14];
    const float* W11  = (const float*)d_in[15];
    const float* b11  = (const float*)d_in[16];
    const float* W21  = (const float*)d_in[17];
    const float* b21  = (const float*)d_in[18];
    const float* W31  = (const float*)d_in[19];
    const float* b31  = (const float*)d_in[20];
    const float* W12  = (const float*)d_in[21];
    const float* b12  = (const float*)d_in[22];
    const float* W22  = (const float*)d_in[23];
    const float* b22  = (const float*)d_in[24];
    const float* W32  = (const float*)d_in[25];
    const float* b32  = (const float*)d_in[26];
    const float* tcng = (const float*)d_in[27];
    const float* tcnb = (const float*)d_in[28];
    const float* rsw  = (const float*)d_in[29];
    const float* rsb  = (const float*)d_in[30];
    const float* rsg  = (const float*)d_in[31];
    const float* rsbb = (const float*)d_in[32];

    // workspace carve-up
    char* w = (char*)d_ws;
    size_t off = 0;
    auto alloc = [&](size_t bytes) -> void* {
        void* p = w + off;
        off = (off + bytes + 255) & ~(size_t)255;
        return p;
    };
    double* meanp   = (double*)alloc(400 * 64 * 8);
    double* m2p     = (double*)alloc((size_t)400 * 4096 * 8);
    double* meanx   = (double*)alloc(64 * 8);
    double* M2      = (double*)alloc(4096 * 8);
    double* ystat   = (double*)alloc((size_t)2 * 192 * 32 * 8);
    double* tmstat  = (double*)alloc((size_t)2 * 192 * 32 * 8);
    float*  scp     = (float*)alloc((size_t)8 * 96 * 625 * 4);
    float*  att     = (float*)alloc((size_t)96 * 625 * 4);
    unsigned short* Wdf  = (unsigned short*)alloc((size_t)192 * 192 * 2);
    unsigned short* U    = (unsigned short*)alloc((size_t)11 * 64 * 192 * 2);
    unsigned short* Wdbf = (unsigned short*)alloc((size_t)192 * 64 * 2);
    unsigned short* Wrbf = (unsigned short*)alloc((size_t)192 * 64 * 2);
    float* bdsum     = (float*)alloc(192 * 4);
    float* cb_tap    = (float*)alloc(192 * 4);
    float* s1        = (float*)alloc(192 * 4);
    float* t1        = (float*)alloc(192 * 4);
    float* s3        = (float*)alloc(192 * 4);
    float* t3        = (float*)alloc(192 * 4);
    float* cbias_g   = (float*)alloc(192 * 4);
    float* cbias_fin = (float*)alloc(192 * 4);
    unsigned short* Xp = (unsigned short*)alloc((size_t)NB * COUT * PIX * 2);  // 78.6 MB, pixel-major
    float* ybuf        = (float*)alloc((size_t)NB * COUT * PIX * 4);           // 157 MB, channel-major
    unsigned short* g  = (unsigned short*)alloc((size_t)NB * COUT * PIX * 2);  // 78.6 MB, pixel-major
    (void)ws_size;

    float* tm = (float*)d_out;   // tm staged in d_out, then overwritten in-place by final GEMM

    // 1) x statistics (for analytic down/res BN)
    k_xstats<<<dim3(400), dim3(256), 0, stream>>>(x, meanp, m2p);
    k_xred<<<dim3(17), dim3(256), 0, stream>>>(meanp, m2p, meanx, M2);

    // 2) weight prep (bf16 GEMM weights, fused TCN taps)
    k_prep<<<dim3(12), dim3(256), 0, stream>>>(Wd, bd, W11, b11, W21, b21, W31, b31,
                                               W12, b12, W22, b22, W32, b32,
                                               Wdf, U, bdsum, cb_tap);

    // 3) analytic BN params for down/res, folded into bf16 weights + biases
    k_bn_lin<<<dim3(1), dim3(256), 0, stream>>>(meanx, M2, dnw, dnb, dng, dnbb,
                                                rsw, rsb, rsg, rsbb,
                                                Wdbf, Wrbf, cbias_g, cbias_fin);

    // 4) attention scores -> softmax -> att
    k_att_scores<<<dim3(96, 8), dim3(256), 0, stream>>>(x, Wa, ba, Wb, bbp, scp);
    k_softmax<<<dim3(96), dim3(32), 0, stream>>>(scp, A, PA, att);

    // 5) z' = x @ att  (bf16, pixel-major)
    k_zprime<<<dim3(8, 3, 32), dim3(256), 0, stream>>>(x, att, Xp);

    // 6) y = Wd_flat @ z' + bdsum  (WMMA bf16, K=192, async-staged B) -> f32 channel-major
    k_gemm<1, 0, 0, 0><<<dim3(50, 32), dim3(256), 0, stream>>>(
        Wdf, (const void*)Xp, (long)COUT * PIX, bdsum, nullptr, nullptr, nullptr,
        (void*)ybuf, 192);

    // 7) gcn BN stats + params
    k_stats<<<dim3(192, 32), dim3(256), 0, stream>>>(ybuf, ystat);
    k_bnfin<<<dim3(1), dim3(192), 0, stream>>>(ystat, gcng, gcnb, s1, t1);

    // 8) g = relu(bn(y) + bn(down(x)))  (WMMA K=64, out bf16 pixel-major)
    k_gemm<0, 1, 1, 1><<<dim3(50, 32), dim3(256), 0, stream>>>(
        Wdbf, (const void*)x, (long)CIN * PIX, cbias_g, ybuf, s1, t1, (void*)g, 64);

    // 9) fused TCN taps (async-staged halo panels) -> tm (in d_out)
    k_tcn<<<dim3(50, 32), dim3(256), 0, stream>>>(U, g, cb_tap, tm);

    // 10) tcn BN stats + params
    k_stats<<<dim3(192, 32), dim3(256), 0, stream>>>(tm, tmstat);
    k_bnfin<<<dim3(1), dim3(192), 0, stream>>>(tmstat, tcng, tcnb, s3, t3);

    // 11) out = relu(bn(tm) + bn(res(x)))  (WMMA K=64, in-place on d_out)
    k_gemm<0, 0, 1, 1><<<dim3(50, 32), dim3(256), 0, stream>>>(
        Wrbf, (const void*)x, (long)CIN * PIX, cbias_fin, tm, s3, t3, (void*)d_out, 64);

    (void)in_sizes; (void)n_in; (void)out_size;
}
